// MultiHeadAttention_85487029060275
// MI455X (gfx1250) — matrix operand order repaired and checked by reading, not compiled, not run on hardware
//
#include <hip/hip_runtime.h>
#include <cstdint>

// ---------------------------------------------------------------------------
// Multi-head attention for MI455X (gfx1250), bf16 WMMA pipeline with
// double-buffered GLOBAL_LOAD_ASYNC_TO_LDS_B128 tile staging.
// B=4, S=2048, D=1024, H=16, dk=64.
// ---------------------------------------------------------------------------

typedef __bf16 bf16_t;
typedef __attribute__((ext_vector_type(16))) __bf16 v16bf;
typedef __attribute__((ext_vector_type(8)))  __bf16 v8bf;
typedef __attribute__((ext_vector_type(8)))  float   v8f;
typedef __attribute__((ext_vector_type(4)))  float   v4f;

#define S_LEN 2048
#define NH    16
#define DK    64
#define DM    1024

enum { PERM_NONE = 0, PERM_HEAD = 1, PERM_HEADT = 2 };

__device__ __forceinline__ v8f wmma_bf16(v16bf a, v16bf b, v8f c) {
  return __builtin_amdgcn_wmma_f32_16x16x32_bf16(false, a, false, b,
                                                 (short)0, c, false, false);
}

__device__ __forceinline__ v16bf combine16(v8bf lo, v8bf hi) {
  return __builtin_shufflevector(lo, hi, 0,1,2,3,4,5,6,7,8,9,10,11,12,13,14,15);
}

// LDS byte offset of a __shared__ object: shared aperture occupies addr[63:32],
// so the low 32 bits of the flat address are the LDS offset (ISA §10.2).
__device__ __forceinline__ unsigned lds_off(const void* p) {
  return (unsigned)(uintptr_t)p;
}

// Async 16B copy global -> LDS (per lane). Tracked by ASYNCcnt.
__device__ __forceinline__ void async_copy16(unsigned lds_addr, const void* gptr) {
  asm volatile("global_load_async_to_lds_b128 %0, %1, off"
               :: "v"(lds_addr), "v"(gptr) : "memory");
}

template <int N>
__device__ __forceinline__ void wait_asynccnt() {
  asm volatile("s_wait_asynccnt %0" :: "i"(N) : "memory");
}

// ---------------------------------------------------------------------------
// fp32 -> bf16 elementwise convert (activations). n multiple of 8.
// ---------------------------------------------------------------------------
__global__ __launch_bounds__(256)
void convert_f32_bf16_kernel(const float* __restrict__ src,
                             bf16_t* __restrict__ dst, long n)
{
  long i = ((long)blockIdx.x * 256 + threadIdx.x) * 8;
  if (i + 8 <= n) {
    v4f a = *(const v4f*)(src + i);
    v4f b = *(const v4f*)(src + i + 4);
    v8bf o;
    #pragma unroll
    for (int e = 0; e < 4; ++e) { o[e] = (bf16_t)a[e]; o[4 + e] = (bf16_t)b[e]; }
    *(v8bf*)(dst + i) = o;
  }
}

// ---------------------------------------------------------------------------
// W (fp32 [K][N]) -> Wt (bf16 [N][K]) transpose+convert, 64x64 LDS tiles.
// ---------------------------------------------------------------------------
__global__ __launch_bounds__(256)
void transpose_w_kernel(const float* __restrict__ W, bf16_t* __restrict__ Wt,
                        int N, int K)
{
  __shared__ bf16_t t[64][72];
  const int tid = threadIdx.x;
  const int n0 = blockIdx.x * 64;
  const int k0 = blockIdx.y * 64;
  #pragma unroll
  for (int it = 0; it < 4; ++it) {
    int idx = tid + it * 256;          // 1024 float4 chunks
    int r   = idx >> 4;                // k row 0..63
    int c4  = idx & 15;                // n col/4
    v4f v = *(const v4f*)(W + (long)(k0 + r) * N + n0 + c4 * 4);
    #pragma unroll
    for (int e = 0; e < 4; ++e) t[c4*4 + e][r] = (bf16_t)v[e];
  }
  __syncthreads();
  #pragma unroll
  for (int it = 0; it < 2; ++it) {
    int idx = tid + it * 256;          // 512 v8bf chunks
    int r   = idx >> 3;                // n row 0..63
    int c8  = idx & 7;                 // k col/8
    *(v8bf*)(Wt + (long)(n0 + r) * K + k0 + c8 * 8) = *(v8bf*)&t[r][c8*8];
  }
}

// ---------------------------------------------------------------------------
// GEMM: out = A(MxK bf16) @ Wt(NxK bf16)^T + bias, optional per-head permute.
// Block tile 128x128, K-stage 64, double-buffered async LDS staging.
// 256 threads = 8 waves; wave tile 64(M)x32(N) = 4x2 WMMA accumulators.
// ---------------------------------------------------------------------------
template <int PERM, typename OT>
__global__ __launch_bounds__(256)
void gemm_wmma_kernel(const bf16_t* __restrict__ A, const bf16_t* __restrict__ Wt,
                      const float* __restrict__ bias, OT* __restrict__ out,
                      int M, int N, int K)
{
  __shared__ bf16_t lA[2][128][72];   // [m][k], row stride 144B (16B aligned)
  __shared__ bf16_t lB[2][128][72];   // [n][k]

  const int tid  = threadIdx.x;
  const int wid  = tid >> 5;
  const int lane = tid & 31;
  const int hv   = lane >> 4;
  const int ml   = lane & 15;
  const int wm   = wid & 1;
  const int wn   = wid >> 1;
  const long bm  = (long)blockIdx.x * 128;
  const long bn  = (long)blockIdx.y * 128;

  auto issue = [&](int buf, int k0) {
    #pragma unroll
    for (int it = 0; it < 4; ++it) {
      int idx = tid + it * 256;        // 1024 16B chunks per tile
      int r   = idx >> 3;              // 0..127
      int c8  = idx & 7;               // 0..7 (k/8)
      async_copy16(lds_off(&lA[buf][r][c8*8]), A  + (bm + r) * (long)K + k0 + c8*8);
      async_copy16(lds_off(&lB[buf][r][c8*8]), Wt + (bn + r) * (long)K + k0 + c8*8);
    }
  };

  const v8f vzero = {0.f,0.f,0.f,0.f,0.f,0.f,0.f,0.f};
  v8f acc[4][2];
  #pragma unroll
  for (int s = 0; s < 4; ++s)
    #pragma unroll
    for (int n = 0; n < 2; ++n) acc[s][n] = vzero;

  const int nstage = K / 64;
  issue(0, 0);
  for (int st = 0; st < nstage; ++st) {
    const int buf = st & 1;
    if (st + 1 < nstage) { issue(buf ^ 1, (st + 1) * 64); wait_asynccnt<8>(); }
    else                 { wait_asynccnt<0>(); }
    __syncthreads();

    #pragma unroll
    for (int ks = 0; ks < 2; ++ks) {   // two 32-wide K sub-steps
      v16bf af[4], bfr[2];
      #pragma unroll
      for (int s = 0; s < 4; ++s) {
        int row = wm*64 + s*16 + ml;   // A frag: K(c) = hv*8+c | 16+hv*8+(c-8)
        v8bf lo = *(v8bf*)&lA[buf][row][ks*32 + hv*8];
        v8bf hi = *(v8bf*)&lA[buf][row][ks*32 + 16 + hv*8];
        af[s] = combine16(lo, hi);
      }
      #pragma unroll
      for (int n = 0; n < 2; ++n) {
        int nr = wn*32 + n*16 + ml;    // B frag: K(c) = hv*8+c | 16+hv*8+(c-8)
        v8bf lo = *(v8bf*)&lB[buf][nr][ks*32 + hv*8];
        v8bf hi = *(v8bf*)&lB[buf][nr][ks*32 + 16 + hv*8];
        bfr[n] = combine16(lo, hi);
      }
      #pragma unroll
      for (int s = 0; s < 4; ++s)
        #pragma unroll
        for (int n = 0; n < 2; ++n)
          acc[s][n] = wmma_bf16(af[s], bfr[n], acc[s][n]);
    }
    __syncthreads();
  }

  // ---- epilogue: C/D layout M = r + 8*hv, N = ml ----
  #pragma unroll
  for (int s = 0; s < 4; ++s) {
    #pragma unroll
    for (int n = 0; n < 2; ++n) {
      int col = (int)bn + wn*32 + n*16 + ml;
      float bv = bias[col];
      #pragma unroll
      for (int r = 0; r < 8; ++r) {
        long row = bm + wm*64 + s*16 + r + hv*8;
        float v = acc[s][n][r] + bv;
        if constexpr (PERM == PERM_NONE) {
          out[row * (long)N + col] = (OT)v;
        } else if constexpr (PERM == PERM_HEAD) {
          long b = row >> 11; long srow = row & (S_LEN - 1);
          int  h = col >> 6;  int  d    = col & (DK - 1);
          out[(((b*NH + h) * (long)S_LEN) + srow) * DK + d] = (OT)v;
        } else { // PERM_HEADT: [bh][d][S]
          long b = row >> 11; long srow = row & (S_LEN - 1);
          int  h = col >> 6;  int  d    = col & (DK - 1);
          out[((b*NH + h) * (long)DK + d) * S_LEN + srow] = (OT)v;
        }
      }
    }
  }
}

// ---------------------------------------------------------------------------
// Flash attention: grid (S/64, B*H); block = 128 threads = 4 waves.
// Wave w owns 16 query rows. 64-key tiles, double-buffered async staging.
// qh,kh: [B*H][S][64] bf16 ; vth: [B*H][64][S] bf16 ; ctx: [B][S][D] bf16.
// ---------------------------------------------------------------------------
__global__ __launch_bounds__(128)
void flash_attn_kernel(const bf16_t* __restrict__ qh,
                       const bf16_t* __restrict__ kh,
                       const bf16_t* __restrict__ vth,
                       bf16_t* __restrict__ ctx)
{
  __shared__ bf16_t lK [2][64][72];    // [t][d]  (B^T layout for QK^T)
  __shared__ bf16_t lVt[2][64][72];    // [d][t]  (B^T layout for P*V)
  __shared__ bf16_t lP [4][16][72];    // per-wave P transpose staging

  const int tid  = threadIdx.x;
  const int wid  = tid >> 5;
  const int lane = tid & 31;
  const int hv   = lane >> 4;
  const int ml   = lane & 15;
  const int bh   = blockIdx.y;
  const size_t base = (size_t)bh * S_LEN * DK;
  const int qrow0 = blockIdx.x * 64 + wid * 16;
  const float scale = 0.125f;          // 1/sqrt(64)

  auto issue = [&](int buf, int t0) {
    #pragma unroll
    for (int it = 0; it < 4; ++it) {
      int idx = tid + it * 128;        // 512 16B chunks per tile
      int r   = idx >> 3;              // 0..63
      int c8  = idx & 7;
      async_copy16(lds_off(&lK [buf][r][c8*8]), kh  + base + (size_t)(t0 + r)*DK + c8*8);
      async_copy16(lds_off(&lVt[buf][r][c8*8]), vth + base + (size_t)r*S_LEN + t0 + c8*8);
    }
  };

  // Q fragments for this wave's 16 rows (VGPR-resident for whole kernel)
  v16bf qa[2];
  {
    const bf16_t* qp = qh + base + (size_t)(qrow0 + ml) * DK;
    #pragma unroll
    for (int i = 0; i < 2; ++i) {
      v8bf lo = *(const v8bf*)(qp + i*32 + hv*8);
      v8bf hi = *(const v8bf*)(qp + i*32 + 16 + hv*8);
      qa[i] = combine16(lo, hi);
    }
  }

  const v8f vzero = {0.f,0.f,0.f,0.f,0.f,0.f,0.f,0.f};
  v8f o[4]; float mrow[8], lrow[8];
  #pragma unroll
  for (int d = 0; d < 4; ++d) o[d] = vzero;
  #pragma unroll
  for (int r = 0; r < 8; ++r) { mrow[r] = -1e30f; lrow[r] = 0.f; }

  const int ntile = S_LEN / 64;
  issue(0, 0);
  for (int st = 0; st < ntile; ++st) {
    const int buf = st & 1;
    if (st + 1 < ntile) { issue(buf ^ 1, (st + 1) * 64); wait_asynccnt<8>(); }
    else                { wait_asynccnt<0>(); }
    __syncthreads();

    // ---- S = Q K^T (16 x 64) ----
    v8f sacc[4];
    #pragma unroll
    for (int ni = 0; ni < 4; ++ni) {
      sacc[ni] = vzero;
      int tr = ni*16 + ml;
      v8bf lo0 = *(v8bf*)&lK[buf][tr][hv*8];
      v8bf hi0 = *(v8bf*)&lK[buf][tr][16 + hv*8];
      sacc[ni] = wmma_bf16(qa[0], combine16(lo0, hi0), sacc[ni]);
      v8bf lo1 = *(v8bf*)&lK[buf][tr][32 + hv*8];
      v8bf hi1 = *(v8bf*)&lK[buf][tr][32 + 16 + hv*8];
      sacc[ni] = wmma_bf16(qa[1], combine16(lo1, hi1), sacc[ni]);
    }

    // ---- online softmax (row r of this half) ----
    float pv[4][8];
    #pragma unroll
    for (int r = 0; r < 8; ++r) {
      float mx = -1e30f;
      #pragma unroll
      for (int ni = 0; ni < 4; ++ni) mx = fmaxf(mx, sacc[ni][r]);
      mx *= scale;
      #pragma unroll
      for (int off = 8; off >= 1; off >>= 1)
        mx = fmaxf(mx, __shfl_xor(mx, off, 32));
      float mnew  = fmaxf(mrow[r], mx);
      float alpha = __expf(mrow[r] - mnew);
      mrow[r] = mnew;
      float lsum = 0.f;
      #pragma unroll
      for (int ni = 0; ni < 4; ++ni) {
        float p = __expf(sacc[ni][r] * scale - mnew);
        pv[ni][r] = p;
        lsum += p;
      }
      #pragma unroll
      for (int off = 8; off >= 1; off >>= 1)
        lsum += __shfl_xor(lsum, off, 32);
      lrow[r] = lrow[r] * alpha + lsum;
      #pragma unroll
      for (int d = 0; d < 4; ++d) o[d][r] *= alpha;
    }

    // ---- transpose P (C layout -> A layout) through wave-private LDS ----
    #pragma unroll
    for (int ni = 0; ni < 4; ++ni)
      #pragma unroll
      for (int r = 0; r < 8; ++r)
        lP[wid][r + hv*8][ni*16 + ml] = (bf16_t)pv[ni][r];
    v16bf pa[2];
    #pragma unroll
    for (int i = 0; i < 2; ++i) {
      v8bf lo = *(v8bf*)&lP[wid][ml][i*32 + hv*8];
      v8bf hi = *(v8bf*)&lP[wid][ml][i*32 + 16 + hv*8];
      pa[i] = combine16(lo, hi);
    }

    // ---- O += P V ----
    #pragma unroll
    for (int di = 0; di < 4; ++di) {
      int dr = di*16 + ml;
      v8bf lo0 = *(v8bf*)&lVt[buf][dr][hv*8];
      v8bf hi0 = *(v8bf*)&lVt[buf][dr][16 + hv*8];
      o[di] = wmma_bf16(pa[0], combine16(lo0, hi0), o[di]);
      v8bf lo1 = *(v8bf*)&lVt[buf][dr][32 + hv*8];
      v8bf hi1 = *(v8bf*)&lVt[buf][dr][32 + 16 + hv*8];
      o[di] = wmma_bf16(pa[1], combine16(lo1, hi1), o[di]);
    }
    __syncthreads();
  }

  // ---- epilogue: ctx[b][s][h*64 + d] bf16 ----
  const int b = bh >> 4;
  const int h = bh & 15;
  #pragma unroll
  for (int di = 0; di < 4; ++di) {
    #pragma unroll
    for (int r = 0; r < 8; ++r) {
      int srow = qrow0 + r + hv*8;
      float val = o[di][r] / lrow[r];
      ctx[((size_t)(b * S_LEN + srow)) * DM + h*DK + di*16 + ml] = (bf16_t)val;
    }
  }
}

// ---------------------------------------------------------------------------
extern "C" void kernel_launch(void* const* d_in, const int* in_sizes, int n_in,
                              void* d_out, int out_size, void* d_ws, size_t ws_size,
                              hipStream_t stream) {
  const float* query = (const float*)d_in[0];
  const float* key_  = (const float*)d_in[1];
  const float* value = (const float*)d_in[2];
  const float* Wq    = (const float*)d_in[3];
  const float* bq    = (const float*)d_in[4];
  const float* Wk    = (const float*)d_in[5];
  const float* bk    = (const float*)d_in[6];
  const float* Wv    = (const float*)d_in[7];
  const float* bv    = (const float*)d_in[8];
  const float* Wo    = (const float*)d_in[9];
  const float* bo    = (const float*)d_in[10];
  float* out = (float*)d_out;

  const int  Bn = 4;
  const int  M  = Bn * S_LEN;          // 8192
  const int  N  = DM;                  // 1024
  const int  K  = DM;                  // 1024
  const long nA = (long)M * DM;        // activation elements

  bf16_t* qhb = (bf16_t*)d_ws;         // [B*H][S][64]
  bf16_t* khb = qhb + nA;              // [B*H][S][64]
  bf16_t* vtb = khb + nA;              // [B*H][64][S]
  bf16_t* ctx = vtb + nA;              // [B][S][D]
  bf16_t* act = ctx + nA;              // converted activation (reused 3x)
  bf16_t* wtb = act + nA;              // transposed weight [N][K] (reused 4x)

  dim3 gCvt((unsigned)(nA / 8 / 256));
  dim3 gTrW(N / 64, K / 64);
  dim3 gGemm(M / 128, N / 128);
  dim3 gAttn(S_LEN / 64, Bn * NH);

  // Q projection
  convert_f32_bf16_kernel<<<gCvt, 256, 0, stream>>>(query, act, nA);
  transpose_w_kernel<<<gTrW, 256, 0, stream>>>(Wq, wtb, N, K);
  gemm_wmma_kernel<PERM_HEAD,  bf16_t><<<gGemm, 256, 0, stream>>>(act, wtb, bq, qhb, M, N, K);
  // K projection
  convert_f32_bf16_kernel<<<gCvt, 256, 0, stream>>>(key_, act, nA);
  transpose_w_kernel<<<gTrW, 256, 0, stream>>>(Wk, wtb, N, K);
  gemm_wmma_kernel<PERM_HEAD,  bf16_t><<<gGemm, 256, 0, stream>>>(act, wtb, bk, khb, M, N, K);
  // V projection (transposed per-head output)
  convert_f32_bf16_kernel<<<gCvt, 256, 0, stream>>>(value, act, nA);
  transpose_w_kernel<<<gTrW, 256, 0, stream>>>(Wv, wtb, N, K);
  gemm_wmma_kernel<PERM_HEADT, bf16_t><<<gGemm, 256, 0, stream>>>(act, wtb, bv, vtb, M, N, K);
  // Attention
  flash_attn_kernel<<<gAttn, 128, 0, stream>>>(qhb, khb, vtb, ctx);
  // Output projection
  transpose_w_kernel<<<gTrW, 256, 0, stream>>>(Wo, wtb, N, K);
  gemm_wmma_kernel<PERM_NONE,  float ><<<gGemm, 256, 0, stream>>>(ctx, wtb, bo, out, M, N, K);
}
